// EncoderAtt_403726926097
// MI455X (gfx1250) — matrix operand order repaired and checked by reading, not compiled, not run on hardware
//
#include <hip/hip_runtime.h>
#include <hip/hip_bf16.h>
#include <math.h>

// ---------------------------------------------------------------------------
// DA-RNN input-attention encoder for MI455X (gfx1250, wave32, WMMA).
// B=512 batch, TM1=256 steps, N=128 series, H=64 hidden.
//
// softmax(s_x + s_hc[:,None] + ab) == softmax(s_x): the h/c terms are
// constant along the softmax axis -> attention weights are time-invariant,
// input_weighted is precomputable, and only h @ W_hh^T is sequential.
// ---------------------------------------------------------------------------

typedef __attribute__((ext_vector_type(16))) __bf16 v16bf;
typedef __attribute__((ext_vector_type(8)))  float  v8f;

#define B_   512
#define TM1_ 256
#define N_   128
#define H_   64
#define G4H_ 256   // 4*H

union FragBF { v16bf v; uint4 q[2]; };
union Pack4  { __bf16 h[4]; uint2 u; };

__device__ __forceinline__ v16bf load_frag(const __bf16* p0, const __bf16* p1) {
  FragBF f;
  f.q[0] = *(const uint4*)p0;   // 16B -> global_load_b128 / ds_load_b128
  f.q[1] = *(const uint4*)p1;
  return f.v;
}

// Workgroup split-barrier that publishes only LDS traffic (s_wait_dscnt 0)
// and deliberately does NOT drain LOADcnt, so prefetched global loads stay
// in flight across the barrier (unlike __syncthreads()'s conservative
// s_wait_loadcnt_dscnt 0x0 fence).
__device__ __forceinline__ void wg_barrier_lds() {
  asm volatile(
      "s_wait_dscnt 0x0\n\t"
      "s_barrier_signal -1\n\t"
      "s_barrier_wait -1"
      ::: "memory");
}

// branch-free: correct limits via exp overflow/underflow, no EXEC gymnastics
__device__ __forceinline__ float fast_sigmoid(float x) {
  return 1.f / (1.f + __expf(-x));
}
__device__ __forceinline__ float fast_tanh(float x) {
  return 1.f - 2.f / (__expf(2.f * x) + 1.f);
}

// ---------------------------------------------------------------------------
// Kernel 0: convert LSTM weights fp32 -> bf16 (row-major [j][k]) for WMMA B.
// ---------------------------------------------------------------------------
__global__ void prep_weights(const float* __restrict__ Wih,
                             const float* __restrict__ Whh,
                             __bf16* __restrict__ wih_bf,
                             __bf16* __restrict__ whh_bf) {
  int i = blockIdx.x * blockDim.x + threadIdx.x;
  int stride = gridDim.x * blockDim.x;
  for (int k = i; k < G4H_ * N_; k += stride) wih_bf[k] = (__bf16)Wih[k];
  for (int k = i; k < G4H_ * H_; k += stride) whh_bf[k] = (__bf16)Whh[k];
}

// ---------------------------------------------------------------------------
// Kernel 1: per-batch attention weights + weighted input.
// One block per batch row b; thread n owns driving series n.
// Second pass over input hits L2 (whole tensor is 67MB < 192MB L2).
// ---------------------------------------------------------------------------
__global__ __launch_bounds__(128) void attn_weight(
    const float* __restrict__ in, const float* __restrict__ attn_w,
    float* __restrict__ out_wx, __bf16* __restrict__ wx_bf) {
  const int b = blockIdx.x;
  const int n = threadIdx.x;
  __shared__ float awx[TM1_];
  __shared__ float red[N_];

  for (int t = n; t < TM1_; t += N_) awx[t] = attn_w[2 * H_ + t];
  __syncthreads();

  const float* row = in + (size_t)b * TM1_ * N_;
  float s = 0.f;
  #pragma unroll 4
  for (int t = 0; t < TM1_; ++t) s += row[t * N_ + n] * awx[t];

  // softmax over n (s_hc and attn_b are constant along n -> cancel exactly)
  red[n] = s; __syncthreads();
  for (int off = 64; off > 0; off >>= 1) {
    if (n < off) red[n] = fmaxf(red[n], red[n + off]);
    __syncthreads();
  }
  float mx = red[0]; __syncthreads();
  float e = __expf(s - mx);
  red[n] = e; __syncthreads();
  for (int off = 64; off > 0; off >>= 1) {
    if (n < off) red[n] += red[n + off];
    __syncthreads();
  }
  float a = e / red[0];

  #pragma unroll 4
  for (int t = 0; t < TM1_; ++t) {
    float w = a * row[t * N_ + n];
    out_wx[(size_t)b * TM1_ * N_ + t * N_ + n] = w;
    wx_bf[(size_t)b * TM1_ * N_ + t * N_ + n]  = (__bf16)w;
  }
}

// ---------------------------------------------------------------------------
// Kernel 2: fused LSTM. 16 blocks x 512 threads (16 wave32 waves).
// Block owns 32 batch rows; wave w: M-tile mt=w&1, j-tile pair jp=w>>1.
// Per step, per wave: 2 j-tiles x (4 input-K + 2 recurrent-K) bf16 WMMAs.
// W_ih/W_hh B fragments live in registers (loop-invariant, 12 frags/wave).
// Input A fragments are double-buffered: step t+1's wx loads are issued
// during step t and ride through the LDS-only split barriers, so only the
// LDS h round-trip sits on the 256-step critical path.
// ---------------------------------------------------------------------------
__global__ __launch_bounds__(512, 1) void lstm_enc(
    const __bf16* __restrict__ wx_bf, const __bf16* __restrict__ wih_bf,
    const __bf16* __restrict__ whh_bf, const float* __restrict__ b_ih,
    const float* __restrict__ b_hh, float* __restrict__ out_enc) {
  __shared__ float  gates[32][G4H_ + 4];   // +4 floats pad vs 64-bank LDS
  __shared__ __bf16 hbuf[32][H_ + 8];      // row stride 144B (16B aligned)

  const int tid  = threadIdx.x;
  const int lane = tid & 31;
  const int wave = tid >> 5;
  const int l16  = lane & 15;
  const int hi   = lane >> 4;       // 0: lanes 0-15, 1: lanes 16-31
  const int mt   = wave & 1;        // M tile within 32-row block tile
  const int jp   = wave >> 1;       // j-tile pair 0..7 (covers 4H=256)
  const int bm0  = blockIdx.x * 32; // first batch row of this block

  for (int k = tid; k < 32 * (H_ + 8); k += 512)
    ((__bf16*)hbuf)[k] = (__bf16)0.0f;

  // ---- loop-invariant B fragments + C init (bias) -------------------------
  // B layout (32x16 bf16): lane = column j (l16), lanes 16-31 start at K+8;
  // per lane: halves [c0, c0+8) and [c0+16, c0+24) = 2 x b128.
  v16bf Bi[2][4];   // input weights, K = 128 -> 4 K-tiles
  v16bf Bh[2][2];   // recurrent weights, K = 64 -> 2 K-tiles
  float cinit[2];
  #pragma unroll
  for (int jl = 0; jl < 2; ++jl) {
    int j = (jp * 2 + jl) * 16 + l16;
    cinit[jl] = b_ih[j] + b_hh[j];
    #pragma unroll
    for (int kt = 0; kt < 4; ++kt) {
      const __bf16* p = wih_bf + j * N_ + kt * 32 + hi * 8;
      Bi[jl][kt] = load_frag(p, p + 16);
    }
    #pragma unroll
    for (int kt = 0; kt < 2; ++kt) {
      const __bf16* p = whh_bf + j * H_ + kt * 32 + hi * 8;
      Bh[jl][kt] = load_frag(p, p + 16);
    }
  }

  // per-thread cell state: row m = tid>>4, cols ec..ec+3
  const int em = tid >> 4;
  const int ec = (tid & 15) * 4;
  float cst[4] = {0.f, 0.f, 0.f, 0.f};

  // ---- A-fragment double buffer -------------------------------------------
  // A layout (16x32 bf16): lane = row M (l16); lanes 16-31 start at K+8;
  // per lane: halves [c0, c0+8) and [c0+16, c0+24)  -> 2 x b128.
  const int am = mt * 16 + l16;
  const __bf16* abase = wx_bf + (size_t)(bm0 + am) * TM1_ * N_;

  v16bf Acur[4], Anxt[4];
  #pragma unroll
  for (int kt = 0; kt < 4; ++kt) {
    int c0 = kt * 32 + hi * 8;
    Acur[kt] = load_frag(abase + c0, abase + c0 + 16);
  }

  wg_barrier_lds();  // hbuf zeroed

  #pragma unroll 2   // makes the Anxt->Acur rotation pure SSA renaming
  for (int t = 0; t < TM1_; ++t) {
    // prefetch next step's input A fragments (independent of h)
    if (t + 1 < TM1_) {
      const __bf16* arow = abase + (size_t)(t + 1) * N_;
      #pragma unroll
      for (int kt = 0; kt < 4; ++kt) {
        int c0 = kt * 32 + hi * 8;
        Anxt[kt] = load_frag(arow + c0, arow + c0 + 16);
      }
    }

    // recurrent A fragments from LDS (previous step's h, bf16)
    v16bf Ah[2];
    #pragma unroll
    for (int kt = 0; kt < 2; ++kt) {
      int c0 = kt * 32 + hi * 8;
      const __bf16* hp = &hbuf[am][0] + c0;
      Ah[kt] = load_frag(hp, hp + 16);
    }

    // ---- gates = bias + wx_t @ Wih^T + h @ Whh^T (bf16 WMMA, f32 acc) -----
    #pragma unroll
    for (int jl = 0; jl < 2; ++jl) {
      float c = cinit[jl];
      v8f acc = {c, c, c, c, c, c, c, c};
      #pragma unroll
      for (int kt = 0; kt < 4; ++kt)
        acc = __builtin_amdgcn_wmma_f32_16x16x32_bf16(
            false, Acur[kt], false, Bi[jl][kt], (short)0, acc, false, false);
      #pragma unroll
      for (int kt = 0; kt < 2; ++kt)
        acc = __builtin_amdgcn_wmma_f32_16x16x32_bf16(
            false, Ah[kt], false, Bh[jl][kt], (short)0, acc, false, false);
      // C/D layout: lane = column (l16), VGPR v = row v (+8 for lanes 16-31)
      int j = (jp * 2 + jl) * 16 + l16;
      #pragma unroll
      for (int v = 0; v < 8; ++v)
        gates[mt * 16 + v + hi * 8][j] = acc[v];
    }
    wg_barrier_lds();  // publish gates; A-prefetch stays in flight

    // ---- elementwise LSTM cell update (gate order i, f, g, o) -------------
    {
      float4 ig = *(const float4*)&gates[em][ec];
      float4 fg = *(const float4*)&gates[em][64 + ec];
      float4 gg = *(const float4*)&gates[em][128 + ec];
      float4 og = *(const float4*)&gates[em][192 + ec];
      float hv[4];
      const float* ip = &ig.x; const float* fp = &fg.x;
      const float* gp = &gg.x; const float* op = &og.x;
      #pragma unroll
      for (int q = 0; q < 4; ++q) {
        float si = fast_sigmoid(ip[q]);
        float sf = fast_sigmoid(fp[q]);
        float so = fast_sigmoid(op[q]);
        float tg = fast_tanh(gp[q]);
        cst[q] = sf * cst[q] + si * tg;
        hv[q]  = so * fast_tanh(cst[q]);
      }
      float4 o4 = {hv[0], hv[1], hv[2], hv[3]};
      *(float4*)(out_enc + ((size_t)(bm0 + em) * TM1_ + t) * H_ + ec) = o4;
      Pack4 pk;
      #pragma unroll
      for (int q = 0; q < 4; ++q) pk.h[q] = (__bf16)hv[q];
      *(uint2*)&hbuf[em][ec] = pk.u;
    }
    wg_barrier_lds();  // publish hbuf for next step's Ah fragments

    // rotate double buffer (eliminated by unroll-2 SSA renaming)
    #pragma unroll
    for (int kt = 0; kt < 4; ++kt) Acur[kt] = Anxt[kt];
  }
}

// ---------------------------------------------------------------------------
extern "C" void kernel_launch(void* const* d_in, const int* in_sizes, int n_in,
                              void* d_out, int out_size, void* d_ws, size_t ws_size,
                              hipStream_t stream) {
  const float* input  = (const float*)d_in[0];  // (B, TM1, N)
  const float* attn_w = (const float*)d_in[1];  // (1, 2H+TM1)
  // d_in[2] = attn_b: constant along softmax axis -> cancels exactly
  const float* W_ih = (const float*)d_in[3];    // (4H, N)
  const float* W_hh = (const float*)d_in[4];    // (4H, H)
  const float* b_ih = (const float*)d_in[5];    // (4H,)
  const float* b_hh = (const float*)d_in[6];    // (4H,)

  float* out_wx  = (float*)d_out;                          // input_weighted
  float* out_enc = (float*)d_out + (size_t)B_ * TM1_ * N_; // input_encoded

  __bf16* wx_bf  = (__bf16*)d_ws;                          // 33.55 MB
  __bf16* wih_bf = wx_bf + (size_t)B_ * TM1_ * N_;         // 64 KB
  __bf16* whh_bf = wih_bf + (size_t)G4H_ * N_;             // 32 KB

  prep_weights<<<64, 256, 0, stream>>>(W_ih, W_hh, wih_bf, whh_bf);
  attn_weight<<<B_, 128, 0, stream>>>(input, attn_w, out_wx, wx_bf);
  lstm_enc<<<B_ / 32, 512, 0, stream>>>(wx_bf, wih_bf, whh_bf, b_ih, b_hh, out_enc);
}
